// InterAgg_28673201668650
// MI455X (gfx1250) — matrix operand order repaired and checked by reading, not compiled, not run on hardware
//
#include <hip/hip_runtime.h>

typedef __attribute__((ext_vector_type(16))) _Float16 v16h;
typedef __attribute__((ext_vector_type(8)))  _Float16 v8h;
typedef __attribute__((ext_vector_type(4)))  _Float16 v4h;
typedef __attribute__((ext_vector_type(8)))  float    v8f;
typedef __attribute__((ext_vector_type(4)))  float    v4f;

#define FEAT 256
#define EMB  256
#define WPAD 264   // padded LDS row stride in halves: dword stride 132 = 4 mod 64 banks

// ---------------------------------------------------------------------------
// Kernel 1: fused  h = X @ W^T  (f16 WMMA, f32 accum)  +  e = h @ a_half
// Block: 256 threads (8 waves) covers 64 rows x 256 cols.
//   wave wv: m-tile = wv>>1 (16 rows), column half = wv&1 (8 N-tiles = 128 cols)
// W staged in LDS as f16 with padded row stride (conflict-free B loads).
// B tiles are software double-buffered so the t+1 LDS loads overlap WMMA t.
// ---------------------------------------------------------------------------
__global__ void interagg_gemm_wmma_kernel(const float* __restrict__ selfF,
                                          const float* __restrict__ neighF,
                                          const float* __restrict__ weight,
                                          const float* __restrict__ avec,
                                          _Float16* __restrict__ h16,
                                          float* __restrict__ e_c,
                                          float* __restrict__ e_n,
                                          int N)
{
    extern __shared__ _Float16 wlds[];        // 256 * WPAD f16 = 132 KB
    __shared__ float epart[8][16];            // per-wave half-row e partials

    const int tid = threadIdx.x;

    // Stage W (row-major [emb][feat]) into LDS as f16, padded rows (vectorized).
    for (int i = tid; i < (EMB * FEAT) / 4; i += 256) {
        const int idx = i * 4;
        const int row = idx >> 8;
        const int col = idx & 255;
        v4f w4 = *(const v4f*)(weight + idx);
        v4h wh;
#pragma unroll
        for (int j = 0; j < 4; ++j) wh[j] = (_Float16)w4[j];
        *(v4h*)(wlds + row * WPAD + col) = wh;
    }
    __syncthreads();

    const int wv      = tid >> 5;
    const int lane    = tid & 31;
    const int mtLocal = wv >> 1;          // 0..3
    const int chalf   = wv & 1;           // 0..1
    const int nBase   = chalf * 8;        // first N-tile index for this wave
    const int m0      = blockIdx.x * 64 + mtLocal * 16;

    const bool isC = (m0 < N);            // wave-uniform: N % 16 == 0
    const float* src = isC ? (selfF + (size_t)m0 * FEAT)
                           : (neighF + (size_t)(m0 - N) * FEAT);

    // A-matrix lane addressing (ISA 16-bit A 16x32 layout):
    //   lane L -> row m0+(L&15); halves 0..7 = K base+0..7, 8..15 = K base+16..23,
    //   base = (L>>4)*8
    const int rA = lane & 15;
    const int kA = (lane >> 4) * 8;
    const float* aptr = src + (size_t)rA * FEAT;

    // B-matrix lane addressing (32x16 tile, B[k][n] = W[n][k]):
    //   lane L -> column n0+(L&15); halves 0..7 = K base+0..7, 8..15 = K base+16..23,
    //   base = k0 + (L>>4)*8
    const _Float16* bbase = wlds + (size_t)(nBase * 16 + (lane & 15)) * WPAD
                                 + (lane >> 4) * 8;

    v8f acc[8] = {};

#pragma unroll
    for (int kk = 0; kk < FEAT / 32; ++kk) {
        const int k0 = kk * 32;
        v4f f0 = *(const v4f*)(aptr + k0 + kA);
        v4f f1 = *(const v4f*)(aptr + k0 + kA + 4);
        v4f f2 = *(const v4f*)(aptr + k0 + kA + 16);
        v4f f3 = *(const v4f*)(aptr + k0 + kA + 20);
        v16h A;
#pragma unroll
        for (int j = 0; j < 4; ++j) {
            A[j]      = (_Float16)f0[j];
            A[4 + j]  = (_Float16)f1[j];
            A[8 + j]  = (_Float16)f2[j];
            A[12 + j] = (_Float16)f3[j];
        }
        // double-buffered B tiles
        const _Float16* bk = bbase + k0;
        v8h b0 = *(const v8h*)(bk);
        v8h b1 = *(const v8h*)(bk + 16);
#pragma unroll
        for (int t = 0; t < 8; ++t) {
            v16h B;
#pragma unroll
            for (int j = 0; j < 8; ++j) { B[j] = b0[j]; B[8 + j] = b1[j]; }
            if (t < 7) {
                const _Float16* bn = bk + (size_t)(t + 1) * 16 * WPAD;
                b0 = *(const v8h*)(bn);
                b1 = *(const v8h*)(bn + 16);
            }
            acc[t] = __builtin_amdgcn_wmma_f32_16x16x32_f16(
                false, A, false, B, (short)0, acc[t], false, false);
        }
    }

    // Store h as f16.  D layout: VGPR j, lane L -> row j + (L<16?0:8), col (L&15).
    const int mOff = (lane < 16) ? 0 : 8;
    const int col  = lane & 15;
#pragma unroll
    for (int t = 0; t < 8; ++t)
#pragma unroll
        for (int j = 0; j < 8; ++j)
            h16[(size_t)(m0 + mOff + j) * EMB + (nBase + t) * 16 + col]
                = (_Float16)acc[t][j];

    // e = h . a_half, this wave's 128-column partial.
    const float* asel = avec + (isC ? 0 : EMB);
    float av[8];
#pragma unroll
    for (int t = 0; t < 8; ++t) av[t] = asel[(nBase + t) * 16 + (lane & 15)];

    float ep[8];
#pragma unroll
    for (int j = 0; j < 8; ++j) ep[j] = 0.f;
#pragma unroll
    for (int t = 0; t < 8; ++t)
#pragma unroll
        for (int j = 0; j < 8; ++j)
            ep[j] += acc[t][j] * av[t];
#pragma unroll
    for (int j = 0; j < 8; ++j) {
        ep[j] += __shfl_xor(ep[j], 1, 32);
        ep[j] += __shfl_xor(ep[j], 2, 32);
        ep[j] += __shfl_xor(ep[j], 4, 32);
        ep[j] += __shfl_xor(ep[j], 8, 32);
    }
    if (lane == 0) {
#pragma unroll
        for (int j = 0; j < 8; ++j) epart[wv][j] = ep[j];       // rows 0..7 of tile
    } else if (lane == 16) {
#pragma unroll
        for (int j = 0; j < 8; ++j) epart[wv][8 + j] = ep[j];   // rows 8..15
    }
    __syncthreads();

    // Combine the two column-half partials; per-row center/neigh routing
    // (a 64-row block may straddle the N boundary).
    if (tid < 64) {
        const int mt = tid >> 4;
        const int rr = tid & 15;
        const float ev = epart[mt * 2 + 0][rr] + epart[mt * 2 + 1][rr];
        const int grow = blockIdx.x * 64 + tid;
        if (grow < N) e_c[grow] = ev;
        else          e_n[grow - N] = ev;
    }
}

// ---------------------------------------------------------------------------
// Kernel 2: per-node softmax attention over R relations, aggregation,
// relu(center+agg) stored transposed as out[d*N + n]. Deterministic per-block
// attention-weight partial sums written to blockAtt[block*R + r].
// Block: 256 threads = 8 warps, handles 32 nodes.
// ---------------------------------------------------------------------------
__global__ void interagg_attn_agg_kernel(const _Float16* __restrict__ h16,
                                         const float* __restrict__ e_c,
                                         const float* __restrict__ e_n,
                                         float* __restrict__ out,
                                         float* __restrict__ blockAtt,
                                         int N, int R)
{
    __shared__ float comb[32][257];   // padded: phase-2 column reads conflict-free
    __shared__ float warpAtt[8][8];

    const int tid = threadIdx.x;
    const int lane = tid & 31;
    const int w = tid >> 5;
    const int n0 = blockIdx.x * 32;

    float attAcc[8];
    for (int r = 0; r < R; ++r) attAcc[r] = 0.f;

    for (int i = 0; i < 4; ++i) {
        const int nl = w * 4 + i;
        const int n = n0 + nl;

        // softmax over R (all lanes redundantly; uniform)
        float x[8];
        const float ec = e_c[n];
        float mx = -1e30f;
        for (int r = 0; r < R; ++r) {
            float v = ec + e_n[(size_t)r * N + n];
            v = v > 0.f ? v : 0.2f * v;      // leaky_relu(0.2)
            x[r] = v;
            mx = fmaxf(mx, v);
        }
        float s = 0.f;
        for (int r = 0; r < R; ++r) { x[r] = __expf(x[r] - mx); s += x[r]; }
        const float inv = 1.f / s;
        for (int r = 0; r < R; ++r) { x[r] *= inv; attAcc[r] += x[r]; }

        // aggregate 8 embedding dims per lane
        const int d0 = lane * 8;
        float o[8];
        v8h ch = *(const v8h*)(h16 + (size_t)n * EMB + d0);
#pragma unroll
        for (int j = 0; j < 8; ++j) o[j] = (float)ch[j];
        for (int r = 0; r < R; ++r) {
            v8h nh = *(const v8h*)(h16 + ((size_t)N + (size_t)r * N + n) * EMB + d0);
            const float wr = x[r];
#pragma unroll
            for (int j = 0; j < 8; ++j) o[j] += wr * (float)nh[j];
        }
#pragma unroll
        for (int j = 0; j < 8; ++j) comb[nl][d0 + j] = fmaxf(o[j], 0.f);
    }

    if (lane == 0)
        for (int r = 0; r < R; ++r) warpAtt[w][r] = attAcc[r];
    __syncthreads();

    // transposed, coalesced store: consecutive lanes -> consecutive n
    const int dBase = w * 32;
    for (int dd = 0; dd < 32; ++dd) {
        const int d = dBase + dd;
        out[(size_t)d * N + n0 + lane] = comb[lane][d];
    }

    if (tid == 0) {
        for (int r = 0; r < R; ++r) {
            float sB = 0.f;
            for (int ww = 0; ww < 8; ++ww) sB += warpAtt[ww][r];
            blockAtt[(size_t)blockIdx.x * R + r] = sB;
        }
    }
}

// ---------------------------------------------------------------------------
// Kernel 3: deterministic reduction of block partials + softmax over R.
// ---------------------------------------------------------------------------
__global__ void interagg_att_final_kernel(const float* __restrict__ blockAtt,
                                          int nBlocks, int R,
                                          float* __restrict__ outAtt)
{
    __shared__ float red[128][8];
    const int tid = threadIdx.x;
    float s[8];
    for (int r = 0; r < R; ++r) s[r] = 0.f;
    for (int b = tid; b < nBlocks; b += 128)
        for (int r = 0; r < R; ++r) s[r] += blockAtt[(size_t)b * R + r];
    for (int r = 0; r < R; ++r) red[tid][r] = s[r];
    __syncthreads();
    if (tid == 0) {
        float tot[8];
        for (int r = 0; r < R; ++r) tot[r] = 0.f;
        for (int t = 0; t < 128; ++t)
            for (int r = 0; r < R; ++r) tot[r] += red[t][r];
        float mx = -1e30f;
        for (int r = 0; r < R; ++r) mx = fmaxf(mx, tot[r]);
        float sum = 0.f;
        float ex[8];
        for (int r = 0; r < R; ++r) { ex[r] = __expf(tot[r] - mx); sum += ex[r]; }
        for (int r = 0; r < R; ++r) outAtt[r] = ex[r] / sum;
    }
}

// ---------------------------------------------------------------------------
extern "C" void kernel_launch(void* const* d_in, const int* in_sizes, int n_in,
                              void* d_out, int out_size, void* d_ws, size_t ws_size,
                              hipStream_t stream)
{
    const float* selfF  = (const float*)d_in[0];
    const float* neighF = (const float*)d_in[1];
    const float* weight = (const float*)d_in[2];
    const float* avec   = (const float*)d_in[3];

    const int N = in_sizes[0] / FEAT;          // 100000
    const int R = in_sizes[1] / in_sizes[0];   // 3

    // workspace layout
    _Float16* h16 = (_Float16*)d_ws;                               // (1+R)*N*256 f16
    const size_t hElems = (size_t)(1 + R) * N * EMB;
    float* e_c = (float*)((char*)d_ws + hElems * sizeof(_Float16)); // N f32
    float* e_n = e_c + N;                                           // R*N f32
    float* blockAtt = e_n + (size_t)R * N;                          // (N/32)*R f32

    float* out    = (float*)d_out;               // combined: 256 x N, row-major
    float* outAtt = out + (size_t)EMB * N;       // att: R values

    const int blocks1 = (int)(((size_t)(1 + R) * N) / 64);   // 6250
    hipLaunchKernelGGL(interagg_gemm_wmma_kernel,
                       dim3(blocks1), dim3(256), EMB * WPAD * sizeof(_Float16), stream,
                       selfF, neighF, weight, avec, h16, e_c, e_n, N);

    const int blocks2 = N / 32;                              // 3125
    interagg_attn_agg_kernel<<<dim3(blocks2), dim3(256), 0, stream>>>(
        h16, e_c, e_n, out, blockAtt, N, R);

    interagg_att_final_kernel<<<dim3(1), dim3(128), 0, stream>>>(
        blockAtt, blocks2, R, outAtt);
}
